// DipNet_19808389169419
// MI455X (gfx1250) — matrix operand order repaired and checked by reading, not compiled, not run on hardware
//
#include <hip/hip_runtime.h>
#include <hip/hip_bf16.h>
#include <math.h>

// ---------------- problem sizes ----------------
constexpr int CB = 256, CS = 17, CL = 81;
constexpr int CNB = 16, CE = 120;
constexpr int CP = 60, CSE = 20;          // power/season emb -> mod = 80
constexpr int CH = 200, COE = 80;
constexpr int CV = 13030, CNVAL = 468;
constexpr int CBO = 35, CPO = 40;
constexpr float CEPS = 1e-5f;

// decoder GEMM tilings
constexpr int CXT = 2 * CE + COE;                 // 320
constexpr int KT_OUT = (CH + 31) / 32;            // 7
constexpr int NT_OUT = (CV + 15) / 16;            // 815
constexpr int KT_IH = (CXT + 31) / 32;            // 10
constexpr int KT_HH = (CH + 31) / 32;             // 7
constexpr int KT_G = KT_IH + KT_HH;               // 17 (combined [xt|h] K)
constexpr int NT_G = (4 * CH + 15) / 16;          // 50
constexpr int MT_B = (CB + 15) / 16;              // 16

typedef __attribute__((ext_vector_type(16))) _Float16 v16h;
typedef __attribute__((ext_vector_type(8)))  float    v8f;

// ---------------- WMMA fragment loaders (fp32 -> f16, branchless) ----------------
// A 16x32 f16 layout: lanes 0-15 hold row M=lane, K = 0..15;
// lanes 16-31 hold K = 16..31 (matches B fragments). Loads are unconditional from clamped
// addresses; out-of-range elements zeroed by select -> loads pipeline freely.
static __device__ __forceinline__ v16h load_A_frag(const float* __restrict__ A, int lda,
                                                   int m0, int k0, int Mlim, int Klim,
                                                   int lane) {
  v16h f;
  int m = m0 + (lane & 15);
  bool mv = (m < Mlim);
  int mc = mv ? m : (Mlim - 1);
  int kb = k0 + ((lane >> 4) << 4);
  const float* row = A + (long)mc * lda;
#pragma unroll
  for (int h = 0; h < 16; ++h) {
    int k = kb + h;
    bool kv = (k < Klim);
    int kc = kv ? k : (Klim - 1);
    float x = row[kc];                       // always in-bounds
    f[h] = (_Float16)((mv && kv) ? x : 0.0f);
  }
  return f;
}

// B 32x16 f16 layout: lane holds column N=lane&15; halves K=0..15 (lanes 0-15)
// or K=16..31 (lanes 16-31). Source stored [K,N] row-major.
static __device__ __forceinline__ v16h load_B_frag_kn(const float* __restrict__ Bm, int ldb,
                                                      int k0, int n0, int Klim, int Nlim,
                                                      int lane) {
  v16h f;
  int n = n0 + (lane & 15);
  bool nv = (n < Nlim);
  int nc = nv ? n : (Nlim - 1);
  int kb = k0 + ((lane >> 4) << 4);
#pragma unroll
  for (int h = 0; h < 16; ++h) {
    int k = kb + h;
    bool kv = (k < Klim);
    int kc = kv ? k : (Klim - 1);
    float x = Bm[(long)kc * ldb + nc];
    f[h] = (_Float16)((nv && kv) ? x : 0.0f);
  }
  return f;
}

// Same fragment, source stored [N,K] row-major (computing A @ W^T).
static __device__ __forceinline__ v16h load_B_frag_nk(const float* __restrict__ Bm, int ldb,
                                                      int k0, int n0, int Klim, int Nlim,
                                                      int lane) {
  v16h f;
  int n = n0 + (lane & 15);
  bool nv = (n < Nlim);
  int nc = nv ? n : (Nlim - 1);
  int kb = k0 + ((lane >> 4) << 4);
  const float* row = Bm + (long)nc * ldb;
#pragma unroll
  for (int h = 0; h < 16; ++h) {
    int k = kb + h;
    bool kv = (k < Klim);
    int kc = kv ? k : (Klim - 1);
    float x = row[kc];
    f[h] = (_Float16)((nv && kv) ? x : 0.0f);
  }
  return f;
}

// ---------------- fragment pack kernels (fp32 -> fragment-ready f16) ----------------
// Packed B layout: out[((kt+ktOff)*NT + nt)*32 + lane] = v16h (32B/lane, b128 loads).
__global__ void k_pack_B_kn(const float* __restrict__ B, int ldb, int K, int N,
                            v16h* __restrict__ out, int NT, int ktOff) {
  int nt = blockIdx.x, kt = blockIdx.y, lane = threadIdx.x;
  out[((long)(kt + ktOff) * NT + nt) * 32 + lane] =
      load_B_frag_kn(B, ldb, kt * 32, nt * 16, K, N, lane);
}

__global__ void k_pack_B_nk(const float* __restrict__ B, int ldb, int K, int N,
                            v16h* __restrict__ out, int NT, int ktOff) {
  int nt = blockIdx.x, kt = blockIdx.y, lane = threadIdx.x;
  out[((long)(kt + ktOff) * NT + nt) * 32 + lane] =
      load_B_frag_nk(B, ldb, kt * 32, nt * 16, K, N, lane);
}

// Packed A layout: out[((mt*KTtot + kt+ktOff)*32 + lane)] = v16h
__global__ void k_pack_A(const float* __restrict__ A, int lda, int M, int K,
                         v16h* __restrict__ out, int KTtot, int ktOff) {
  int mt = blockIdx.x, kt = blockIdx.y, lane = threadIdx.x;
  out[((long)mt * KTtot + kt + ktOff) * 32 + lane] =
      load_A_frag(A, lda, mt * 16, kt * 32, M, K, lane);
}

// ---------------- generic batched WMMA GEMM (encoder path) ----------------
// C[M,N] = A[M,K] * B[K,N] + bias.  Grid: x=ceil(NT/8), y=ceil(M/16), z=batch.
__global__ void wmma_gemm_f32(const float* __restrict__ A, long sAb, int lda,
                              const float* __restrict__ Bm, long sBb, int ldb,
                              const float* __restrict__ bias,
                              float* __restrict__ C, long sCb, int ldc,
                              int M, int N, int K) {
  int lane = threadIdx.x & 31;
  int wave = threadIdx.x >> 5;
  int n0 = (blockIdx.x * 8 + wave) * 16;
  if (n0 >= N) return;                 // wave-uniform exit
  int m0 = blockIdx.y * 16;
  const float* Ab = A + (long)blockIdx.z * sAb;
  const float* Bb = Bm + (long)blockIdx.z * sBb;
  float* Cb = C + (long)blockIdx.z * sCb;

  v8f acc = {};
  for (int k0 = 0; k0 < K; k0 += 32) {
    v16h a = load_A_frag(Ab, lda, m0, k0, M, K, lane);
    v16h b = load_B_frag_kn(Bb, ldb, k0, n0, K, N, lane);
    acc = __builtin_amdgcn_wmma_f32_16x16x32_f16(false, a, false, b, (short)0, acc,
                                                 false, false);
  }
  int n = n0 + (lane & 15);
  if (n >= N) return;
  float bv = bias ? bias[n] : 0.0f;
  int mb = m0 + ((lane >> 4) << 3);
#pragma unroll
  for (int r = 0; r < 8; ++r) {
    int m = mb + r;
    if (m < M) Cb[(long)m * ldc + n] = acc[r] + bv;
  }
}

// FiLM coefficients, both gamma and beta in one launch (blockIdx.z selects).
__global__ void wmma_film(const float* __restrict__ mod,
                          const float* __restrict__ gW, const float* __restrict__ gb,
                          const float* __restrict__ bW, const float* __restrict__ bb,
                          float* __restrict__ gamma, float* __restrict__ beta) {
  int lane = threadIdx.x & 31;
  int wave = threadIdx.x >> 5;
  int n0 = wave * 16;
  if (n0 >= CE) return;
  int m0 = blockIdx.y * 16;
  const float* Bm = blockIdx.z ? bW : gW;
  const float* bias = blockIdx.z ? bb : gb;
  float* C = blockIdx.z ? beta : gamma;

  v8f acc = {};
  for (int k0 = 0; k0 < CP + CSE; k0 += 32) {
    v16h a = load_A_frag(mod, CP + CSE, m0, k0, CB, CP + CSE, lane);
    v16h b = load_B_frag_kn(Bm, CE, k0, n0, CP + CSE, CE, lane);
    acc = __builtin_amdgcn_wmma_f32_16x16x32_f16(false, a, false, b, (short)0, acc,
                                                 false, false);
  }
  int n = n0 + (lane & 15);
  float bv = bias[n];
  int mb = m0 + ((lane >> 4) << 3);
#pragma unroll
  for (int r = 0; r < 8; ++r) C[(long)(mb + r) * CE + n] = acc[r] + bv;
}

// ---------------- packed-operand WMMA GEMM (combined LSTM gates) ----------------
__global__ void wmma_gemm_packed(const v16h* __restrict__ Ap, int KT,
                                 const v16h* __restrict__ Bp, int NT,
                                 const float* __restrict__ bias,
                                 float* __restrict__ C, int ldc, int M, int N) {
  int lane = threadIdx.x & 31;
  int wave = threadIdx.x >> 5;
  int nt = blockIdx.x * 8 + wave;
  if (nt >= NT) return;
  int mt = blockIdx.y;

  v8f acc = {};
  const v16h* ap = Ap + ((long)mt * KT) * 32 + lane;
  const v16h* bp = Bp + (long)nt * 32 + lane;
  for (int kt = 0; kt < KT; ++kt) {
    v16h a = ap[(long)kt * 32];
    v16h b = bp[(long)kt * NT * 32];
    acc = __builtin_amdgcn_wmma_f32_16x16x32_f16(false, a, false, b, (short)0, acc,
                                                 false, false);
  }
  int n = nt * 16 + (lane & 15);
  if (n >= N) return;
  float bv = bias ? bias[n] : 0.0f;
  int mb = mt * 16 + ((lane >> 4) << 3);
#pragma unroll
  for (int r = 0; r < 8; ++r) {
    int m = mb + r;
    if (m < M) C[(long)m * ldc + n] = acc[r] + bv;
  }
}

// ---------------- fused scores GEMM: h@W_out + b_out, mask, write [B,S,V] -------------
__global__ void wmma_scores(const v16h* __restrict__ hP,     // packed A [MT][KT][32]
                            const v16h* __restrict__ WoutP,  // packed B [KT][NT][32]
                            const float* __restrict__ bout,  // [V]
                            const unsigned char* __restrict__ mask, // [B,V] stamps
                            const int* __restrict__ loc_idxs,       // [B,S]
                            const int* __restrict__ teacher,        // [B,S]
                            float* __restrict__ out_scores,         // [B,S,V]
                            int t) {
  int lane = threadIdx.x & 31;
  int wave = threadIdx.x >> 5;
  int nt = blockIdx.x * 8 + wave;
  if (nt >= NT_OUT) return;
  int mt = blockIdx.y;

  v8f acc = {};
  const v16h* ap = hP + ((long)mt * KT_OUT) * 32 + lane;
  const v16h* bp = WoutP + (long)nt * 32 + lane;
#pragma unroll
  for (int kt = 0; kt < KT_OUT; ++kt) {
    v16h a = ap[(long)kt * 32];
    v16h b = bp[(long)kt * NT_OUT * 32];
    acc = __builtin_amdgcn_wmma_f32_16x16x32_f16(false, a, false, b, (short)0, acc,
                                                 false, false);
  }
  int v = nt * 16 + (lane & 15);
  if (v >= CV) return;
  float bv = bout[v];
  unsigned char stamp = (unsigned char)(t + 1);
  int mb = mt * 16 + ((lane >> 4) << 3);
#pragma unroll
  for (int r = 0; r < 8; ++r) {
    int b = mb + r;
    float sc = acc[r] + bv;
    int loc = loc_idxs[b * CS + t];
    bool ok;
    if (loc == -1) {
      ok = true;                                   // invalid step: no masking
    } else {
      ok = (v != 0) && (mask[(long)b * CV + v] == stamp);
      if (ok) {
        for (int s = 0; s < t; ++s) {              // cumulative (compatible-orders) mask
          if (teacher[b * CS + s] == v) { ok = false; break; }
        }
      }
    }
    if (!ok) sc -= 1e9f;
    out_scores[((long)b * CS + t) * CV + v] = sc;
  }
}

// ---------------- small VALU kernels ----------------
__global__ void k_mod(const float* __restrict__ p1h, const float* __restrict__ s1h,
                      const float* __restrict__ pW, const float* __restrict__ pb,
                      const float* __restrict__ sW, const float* __restrict__ sb,
                      float* __restrict__ mod) {
  int b = blockIdx.x, tid = threadIdx.x;
  if (tid < CP) {
    float a = pb[tid];
    for (int k = 0; k < 7; ++k) a += p1h[b * 7 + k] * pW[k * CP + tid];
    mod[b * (CP + CSE) + tid] = a;
  } else if (tid < CP + CSE) {
    int j = tid - CP;
    float a = sb[j];
    for (int k = 0; k < 3; ++k) a += s1h[b * 3 + k] * sW[k * CSE + j];
    mod[b * (CP + CSE) + tid] = a;
  }
}

__global__ void k_bn_stats(const float* __restrict__ y, float* __restrict__ mu,
                           float* __restrict__ rstd) {
  int l = blockIdx.x, tid = threadIdx.x;
  float s = 0.f, s2 = 0.f;
  for (int i = tid; i < CB * CE; i += 256) {
    int b = i / CE, e = i - b * CE;
    float v = y[((long)b * CL + l) * CE + e];
    s += v; s2 += v * v;
  }
  __shared__ float sh[256], sh2[256];
  sh[tid] = s; sh2[tid] = s2; __syncthreads();
  for (int st = 128; st > 0; st >>= 1) {
    if (tid < st) { sh[tid] += sh[tid + st]; sh2[tid] += sh2[tid + st]; }
    __syncthreads();
  }
  if (tid == 0) {
    float m = sh[0] / (float)(CB * CE);
    float var = sh2[0] / (float)(CB * CE) - m * m;
    mu[l] = m;
    rstd[l] = rsqrtf(var + CEPS);
  }
}

__global__ void k_bn_film(const float* __restrict__ y, const float* __restrict__ mu,
                          const float* __restrict__ rstd, const float* __restrict__ bn_g,
                          const float* __restrict__ bn_b, const float* __restrict__ gamma,
                          const float* __restrict__ beta, const float* __restrict__ xin,
                          float* __restrict__ xout, int residual, long n) {
  long i = (long)blockIdx.x * 256 + threadIdx.x;
  if (i >= n) return;
  int e = (int)(i % CE);
  long bl = i / CE;
  int l = (int)(bl % CL);
  int b = (int)(bl / CL);
  float v = (y[i] - mu[l]) * rstd[l] * bn_g[l] + bn_b[l];
  v = gamma[b * CE + e] * v + beta[b * CE + e];
  v = fmaxf(v, 0.0f);
  if (residual) v += xin[i];
  xout[i] = v;
}

__global__ void k_concat(const float* __restrict__ x, float* __restrict__ enc, int off) {
  long i = (long)blockIdx.x * 256 + threadIdx.x;
  if (i >= (long)CB * CL * CE) return;
  long bl = i / CE;
  int e = (int)(i % CE);
  enc[bl * (2 * CE) + off + e] = x[i];
}

__global__ void k_bsum(const float* __restrict__ a, const float* __restrict__ b,
                       float* __restrict__ o) {
  int i = blockIdx.x * 256 + threadIdx.x;
  if (i < 4 * CH) o[i] = a[i] + b[i];
}

__global__ void k_adj_align(const float* __restrict__ MA, const int* __restrict__ loc_idxs,
                            float* __restrict__ aa) {
  int i = blockIdx.x * 256 + threadIdx.x;
  if (i >= CB * CL) return;
  int b = i / CL, l = i - b * CL;
  float s = 0.f;
  for (int t = 0; t < CS; ++t) {
    int loc = loc_idxs[b * CS + t];
    int row = (loc < 0) ? CL : loc;          // row L of MA is zeros
    s += MA[row * CL + l];
  }
  aa[i] = s;
}

__global__ void k_zero_f32(float* __restrict__ p, long n) {
  long i = (long)blockIdx.x * 256 + threadIdx.x;
  if (i < n) p[i] = 0.0f;
}

__global__ void k_zero_u8(unsigned char* __restrict__ p, long n) {
  long i = (long)blockIdx.x * 256 + threadIdx.x;
  if (i < n) p[i] = 0;
}

__global__ void k_scatter_mask(const int* __restrict__ vidx, unsigned char* __restrict__ mask,
                               int t) {
  int i = blockIdx.x * 256 + threadIdx.x;
  if (i >= CB * CNVAL) return;
  int b = i / CNVAL, j = i - b * CNVAL;
  int v = vidx[((long)b * CS + t) * CNVAL + j];
  mask[(long)b * CV + v] = (unsigned char)(t + 1);   // step stamp: no re-zeroing needed
}

// align + loc_enc + xt assembly; also fuses oe = emb[teacher[t-1]] (zeros at t==0)
__global__ void k_align_locenc(const float* __restrict__ MA, const float* __restrict__ aa,
                               const int* __restrict__ in_adj, const int* __restrict__ loc_idxs,
                               const float* __restrict__ enc,
                               const float* __restrict__ emb, const int* __restrict__ teacher,
                               float* __restrict__ xt, int t) {
  __shared__ float al[CL];
  __shared__ float ssum;
  int b = blockIdx.x, tid = threadIdx.x;
  if (tid < CL) {
    int loc = loc_idxs[b * CS + t];
    int row = (loc < 0) ? CL : loc;
    al[tid] = in_adj[b] ? aa[b * CL + tid] : MA[row * CL + tid];
  }
  __syncthreads();
  if (tid == 0) {
    float s = 0.f;
    for (int l = 0; l < CL; ++l) s += al[l];
    ssum = s;
  }
  __syncthreads();
  float s = ssum;
  float inv = (s != 0.0f) ? 1.0f / s : 0.0f;
  for (int e = tid; e < 2 * CE; e += blockDim.x) {
    float a = 0.f;
    for (int l = 0; l < CL; ++l) a += al[l] * enc[((long)b * CL + l) * (2 * CE) + e];
    xt[b * CXT + e] = a * inv;
  }
  for (int j = tid; j < COE; j += blockDim.x) {
    float v = 0.f;
    if (t > 0) {
      int tf = teacher[b * CS + (t - 1)];
      v = emb[(long)tf * COE + j];
    }
    xt[b * CXT + 2 * CE + j] = v;
  }
}

static __device__ __forceinline__ float sigf(float x) { return 1.0f / (1.0f + expf(-x)); }

__global__ void k_lstm_update(const float* __restrict__ gates, float* __restrict__ h,
                              float* __restrict__ c) {
  int i = blockIdx.x * 256 + threadIdx.x;
  if (i >= CB * CH) return;
  int b = i / CH, j = i - b * CH;
  const float* g = gates + (long)b * 4 * CH;
  float ig = sigf(g[j]);
  float fg = sigf(g[CH + j]);
  float gg = tanhf(g[2 * CH + j]);
  float og = sigf(g[3 * CH + j]);
  float cn = fg * c[i] + ig * gg;
  c[i] = cn;
  h[i] = og * tanhf(cn);
}

__global__ void k_argmax(const float* __restrict__ out_scores, float* __restrict__ idx_out,
                         int t) {
  int b = blockIdx.x, tid = threadIdx.x;
  const float* sc = out_scores + ((long)b * CS + t) * CV;
  float best = -3.4e38f;
  int bi = 0x7fffffff;
  for (int v = tid; v < CV; v += 256) {
    float x = sc[v];
    if (x > best || (x == best && v < bi)) { best = x; bi = v; }
  }
  __shared__ float bs[256];
  __shared__ int bidx[256];
  bs[tid] = best; bidx[tid] = bi; __syncthreads();
  for (int st = 128; st > 0; st >>= 1) {
    if (tid < st) {
      if (bs[tid + st] > bs[tid] ||
          (bs[tid + st] == bs[tid] && bidx[tid + st] < bidx[tid])) {
        bs[tid] = bs[tid + st]; bidx[tid] = bidx[tid + st];
      }
    }
    __syncthreads();
  }
  if (tid == 0) idx_out[b * CS + t] = (float)bidx[0];   // first-index tie-break, like argmax
}

// ---------------- host orchestration ----------------
static inline long cdiv(long a, long b) { return (a + b - 1) / b; }

extern "C" void kernel_launch(void* const* d_in, const int* in_sizes, int n_in,
                              void* d_out, int out_size, void* d_ws, size_t ws_size,
                              hipStream_t stream) {
  (void)in_sizes; (void)n_in; (void)out_size; (void)ws_size;

  const float* x_bo     = (const float*)d_in[0];
  const float* x_po     = (const float*)d_in[1];
  const float* x_power  = (const float*)d_in[2];
  const float* x_season = (const float*)d_in[3];
  const float* Aadj     = (const float*)d_in[4];
  const float* power_W  = (const float*)d_in[5];
  const float* power_b  = (const float*)d_in[6];
  const float* season_W = (const float*)d_in[7];
  const float* season_b = (const float*)d_in[8];
  struct StreamP { const float *W0, *b0, *W, *b, *bn_g, *bn_b, *gW, *gb, *bW, *bb; };
  StreamP sp[2];
  for (int s2 = 0; s2 < 2; ++s2) {
    int base = 9 + s2 * 10;
    sp[s2].W0   = (const float*)d_in[base + 0];
    sp[s2].b0   = (const float*)d_in[base + 1];
    sp[s2].W    = (const float*)d_in[base + 2];
    sp[s2].b    = (const float*)d_in[base + 3];
    sp[s2].bn_g = (const float*)d_in[base + 4];
    sp[s2].bn_b = (const float*)d_in[base + 5];
    sp[s2].gW   = (const float*)d_in[base + 6];
    sp[s2].gb   = (const float*)d_in[base + 7];
    sp[s2].bW   = (const float*)d_in[base + 8];
    sp[s2].bb   = (const float*)d_in[base + 9];
  }
  const float* emb   = (const float*)d_in[29];
  const float* W_ih  = (const float*)d_in[30];
  const float* W_hh  = (const float*)d_in[31];
  const float* b_ih  = (const float*)d_in[32];
  const float* b_hh  = (const float*)d_in[33];
  const float* W_out = (const float*)d_in[34];
  const float* b_out = (const float*)d_in[35];
  const float* MA    = (const float*)d_in[36];
  const int* in_adj  = (const int*)d_in[37];
  const int* loc_idxs = (const int*)d_in[38];
  const int* vidx    = (const int*)d_in[39];
  const int* teacher = (const int*)d_in[40];

  float* out = (float*)d_out;
  float* out_scores = out + (long)CB * CS;   // layout: idxs [B,S] then scores [B,S,V]

  // workspace bump allocator (256B aligned blocks)
  char* w = (char*)d_ws;
  auto alloc = [&](size_t bytes) -> char* {
    char* p = w;
    w += (bytes + 255) & ~(size_t)255;
    return p;
  };
  float* enc   = (float*)alloc((size_t)CB * CL * 2 * CE * 4);
  float* xb    = (float*)alloc((size_t)CB * CL * CE * 4);
  float* wx    = (float*)alloc((size_t)CB * CL * CE * 4);
  float* yb    = (float*)alloc((size_t)CB * CL * CE * 4);
  float* modb  = (float*)alloc((size_t)CB * (CP + CSE) * 4);
  float* gamma = (float*)alloc((size_t)CB * CE * 4);
  float* beta  = (float*)alloc((size_t)CB * CE * 4);
  float* mu    = (float*)alloc((size_t)CL * 4);
  float* rstd  = (float*)alloc((size_t)CL * 4);
  float* aa    = (float*)alloc((size_t)CB * CL * 4);
  float* xt    = (float*)alloc((size_t)CB * CXT * 4);
  float* state = (float*)alloc((size_t)CB * 2 * CH * 4);
  float* hb = state;
  float* cb = state + (long)CB * CH;
  float* gates = (float*)alloc((size_t)CB * 4 * CH * 4);
  float* bsum  = (float*)alloc((size_t)4 * CH * 4);
  unsigned char* mask = (unsigned char*)alloc((size_t)CB * CV);
  // packed f16 fragment buffers
  v16h* WoutP = (v16h*)alloc((size_t)KT_OUT * NT_OUT * 32 * sizeof(v16h));
  v16h* WgP   = (v16h*)alloc((size_t)KT_G * NT_G * 32 * sizeof(v16h));   // [W_ih|W_hh]
  v16h* AgP   = (v16h*)alloc((size_t)MT_B * KT_G * 32 * sizeof(v16h));   // [xt|h]
  v16h* hP    = (v16h*)alloc((size_t)MT_B * KT_OUT * 32 * sizeof(v16h));

  // ---- preliminaries ----
  k_mod<<<CB, 128, 0, stream>>>(x_power, x_season, power_W, power_b, season_W, season_b, modb);
  k_bsum<<<cdiv(4 * CH, 256), 256, 0, stream>>>(b_ih, b_hh, bsum);
  k_adj_align<<<cdiv(CB * CL, 256), 256, 0, stream>>>(MA, loc_idxs, aa);
  k_zero_f32<<<cdiv((long)CB * 2 * CH, 256), 256, 0, stream>>>(state, (long)CB * 2 * CH);
  k_zero_u8<<<cdiv((long)CB * CV, 256), 256, 0, stream>>>(mask, (long)CB * CV);
  // one-time weight packing for the decoder (reused across all 17 steps)
  k_pack_B_kn<<<dim3(NT_OUT, KT_OUT), 32, 0, stream>>>(W_out, CV, CH, CV, WoutP, NT_OUT, 0);
  k_pack_B_nk<<<dim3(NT_G, KT_IH), 32, 0, stream>>>(W_ih, CXT, CXT, 4 * CH, WgP, NT_G, 0);
  k_pack_B_nk<<<dim3(NT_G, KT_HH), 32, 0, stream>>>(W_hh, CH, CH, 4 * CH, WgP, NT_G, KT_IH);

  // ---- encoder: two streams x 16 blocks ----
  for (int s2 = 0; s2 < 2; ++s2) {
    const float* xin0 = (s2 == 0) ? x_bo : x_po;
    int Cin0 = (s2 == 0) ? CBO : CPO;
    const StreamP& p = sp[s2];
    for (int i = 0; i < CNB; ++i) {
      const float* Wp = (i == 0) ? p.W0 : p.W + (size_t)(i - 1) * CE * CE;
      const float* bp = (i == 0) ? p.b0 : p.b + (size_t)(i - 1) * CE;
      const float* Ain = (i == 0) ? xin0 : xb;
      int Kd = (i == 0) ? Cin0 : CE;
      // Wx = x @ W + b : [B*L, E]
      dim3 g1(1, (unsigned)cdiv(CB * CL, 16), 1);
      wmma_gemm_f32<<<g1, 256, 0, stream>>>(
          Ain, 0, Kd, Wp, 0, CE, bp, wx, 0, CE, CB * CL, CE, Kd);
      // y[b] = A @ Wx[b] : batched over B
      dim3 g2(1, (unsigned)cdiv(CL, 16), CB);
      wmma_gemm_f32<<<g2, 256, 0, stream>>>(
          Aadj, 0, CL, wx, (long)CL * CE, CE, nullptr, yb, (long)CL * CE, CE, CL, CE, CL);
      // BN statistics per location over (B, E)
      k_bn_stats<<<CL, 256, 0, stream>>>(yb, mu, rstd);
      // FiLM coefficients: gamma/beta = mod @ {gW,bW} + {gb,bb} (one launch, z=2)
      dim3 g3(1, (unsigned)cdiv(CB, 16), 2);
      wmma_film<<<g3, 256, 0, stream>>>(
          modb, p.gW + (size_t)i * (CP + CSE) * CE, p.gb + (size_t)i * CE,
          p.bW + (size_t)i * (CP + CSE) * CE, p.bb + (size_t)i * CE, gamma, beta);
      // BN + FiLM + ReLU + residual
      long nel = (long)CB * CL * CE;
      k_bn_film<<<cdiv(nel, 256), 256, 0, stream>>>(
          yb, mu, rstd, p.bn_g + (size_t)i * CL, p.bn_b + (size_t)i * CL,
          gamma, beta, xb, xb, (i > 0) ? 1 : 0, nel);
    }
    k_concat<<<cdiv((long)CB * CL * CE, 256), 256, 0, stream>>>(xb, enc, s2 * CE);
  }

  // ---- decoder: 17 sequential steps ----
  for (int t = 0; t < CS; ++t) {
    k_scatter_mask<<<cdiv(CB * CNVAL, 256), 256, 0, stream>>>(vidx, mask, t);
    k_align_locenc<<<CB, 256, 0, stream>>>(MA, aa, in_adj, loc_idxs, enc, emb, teacher,
                                           xt, t);
    // pack [xt | h] into combined fragment buffer (K chunks 0..9 | 10..16)
    k_pack_A<<<dim3(MT_B, KT_IH), 32, 0, stream>>>(xt, CXT, CB, CXT, AgP, KT_G, 0);
    k_pack_A<<<dim3(MT_B, KT_HH), 32, 0, stream>>>(hb, CH, CB, CH, AgP, KT_G, KT_IH);
    // gates = [xt|h] @ [W_ih|W_hh]^T + (b_ih + b_hh)  -- single fused GEMM
    dim3 gg((unsigned)cdiv(NT_G, 8), MT_B, 1);
    wmma_gemm_packed<<<gg, 256, 0, stream>>>(AgP, KT_G, WgP, NT_G, bsum,
                                             gates, 4 * CH, CB, 4 * CH);
    k_lstm_update<<<cdiv(CB * CH, 256), 256, 0, stream>>>(gates, hb, cb);
    // pack updated h for the output projection
    k_pack_A<<<dim3(MT_B, KT_OUT), 32, 0, stream>>>(hb, CH, CB, CH, hP, KT_OUT, 0);
    // scores = h @ W_out + b_out, fused masking, written straight to output
    dim3 gs((unsigned)cdiv(NT_OUT, 8), MT_B, 1);
    wmma_scores<<<gs, 256, 0, stream>>>(hP, WoutP, b_out, mask, loc_idxs, teacher,
                                        out_scores, t);
    k_argmax<<<CB, 256, 0, stream>>>(out_scores, out, t);
  }
}
